// DualMultiHeadAttention_38835094291003
// MI455X (gfx1250) — matrix operand order repaired and checked by reading, not compiled, not run on hardware
//
#include <hip/hip_runtime.h>
#include <stdint.h>

// ---------------------------------------------------------------------------
// Types for CDNA5 WMMA (wave32): V_WMMA_F32_16X16X32_BF16
// ---------------------------------------------------------------------------
typedef __attribute__((ext_vector_type(16))) __bf16 v16bf;
typedef __attribute__((ext_vector_type(8)))  float  v8f;

union Frag16 {
  v16bf    v;
  uint32_t u[8];
  uint16_t h[16];
  uint4    q[2];
};

#define WMMA_BF16(A, B, C) \
  __builtin_amdgcn_wmma_f32_16x16x32_bf16(false, (A), false, (B), (short)0, (C), false, false)

#ifndef __has_builtin
#define __has_builtin(x) 0
#endif

// CDNA5 async global->LDS path (ASYNCcnt); fall back to plain copies if the
// builtin is absent so compilation never breaks.
#if defined(__HIP_DEVICE_COMPILE__) && __has_builtin(__builtin_amdgcn_global_load_async_to_lds_b128)
#define HAVE_ASYNC_LDS 1
#else
#define HAVE_ASYNC_LDS 0
#endif

#if HAVE_ASYNC_LDS
typedef int v4i_gcc __attribute__((vector_size(16)));
typedef __attribute__((address_space(1))) v4i_gcc* g_v4i_p;   // global int4*
typedef __attribute__((address_space(3))) v4i_gcc* l_v4i_p;   // LDS int4*
#endif

__device__ __forceinline__ void copy16_g2l(const uint16_t* g, uint16_t* l) {
#if HAVE_ASYNC_LDS
  __builtin_amdgcn_global_load_async_to_lds_b128((g_v4i_p)(g), (l_v4i_p)(l), 0, 0);
#else
  *(uint4*)l = *(const uint4*)g;
#endif
}

__device__ __forceinline__ void wait_async_done() {
#if HAVE_ASYNC_LDS
#if __has_builtin(__builtin_amdgcn_s_wait_asynccnt)
  __builtin_amdgcn_s_wait_asynccnt(0);
#else
  asm volatile("s_wait_asynccnt 0" ::: "memory");
#endif
#endif
}

__device__ __forceinline__ uint16_t f2bf(float f) {
  uint32_t u = __float_as_uint(f);
  u += 0x7FFFu + ((u >> 16) & 1u);       // round-to-nearest-even
  return (uint16_t)(u >> 16);
}

static constexpr int Bv   = 2;
static constexpr int N1   = 1024;
static constexpr int NTOT = 2048;
static constexpr int Dm   = 2048;
static constexpr int Hh   = 16;
static constexpr int DH   = 128;

// ---------------------------------------------------------------------------
// GEMM  [1024 x 2048] x [2048 x 2048]  + fused epilogue
//  EPI 0: Q  -> bias, *DH^-0.5, RoPE, store bf16 [B,H,N,DH]
//  EPI 1: K  -> bias, RoPE,            store bf16 [B,H,N,DH]
//  EPI 2: V  -> bias,                  store bf16 [B,H,DH,N] (transposed)
//  EPI 3: O-proj (A is bf16, async-staged) -> bias, store f32 to d_out
// Block: 256 thr (8 waves), tile 128x128; wave tile 32x64 (2x4 WMMA tiles).
// LDS row stride 40 halfwords: 16B-aligned rows, 20-dword stride is
// bank-conflict-free for the 16-lane fragment read patterns.
// ---------------------------------------------------------------------------
template<int EPI>
__global__ __launch_bounds__(256)
void gemm_epi_kernel(const float* __restrict__ x1, const float* __restrict__ x2,
                     const uint16_t* __restrict__ Abf,
                     const float* __restrict__ w_s0, const float* __restrict__ w_s1,
                     const float* __restrict__ b_s0, const float* __restrict__ b_s1,
                     const float* __restrict__ cosT1, const float* __restrict__ sinT1,
                     const float* __restrict__ cosT2, const float* __restrict__ sinT2,
                     uint16_t* __restrict__ out_q, uint16_t* __restrict__ out_k,
                     uint16_t* __restrict__ out_vt, float* __restrict__ out_f)
{
  __shared__ uint16_t As[128][40];   // [row][k]  bf16
  __shared__ uint16_t Bs[128][40];   // [col][k]  bf16 (W tile transposed)

  const int t      = threadIdx.x;
  const int bs     = blockIdx.y;
  const int b      = bs >> 1, stream = bs & 1;
  const int tileM  = blockIdx.x & 7, tileN = blockIdx.x >> 3;
  const int rowBase = tileM * 128, colBase = tileN * 128;
  const int wid = t >> 5, lane = t & 31, half = lane >> 4, l15 = lane & 15;
  const int wr = (wid & 3) * 32, wc = (wid >> 2) * 64;

  const float*    Ap = nullptr;
  const uint16_t* Ab = nullptr;
  if (EPI < 3) Ap = (stream ? x2 : x1) + (size_t)b * N1 * Dm;
  else         Ab = Abf + ((size_t)b * NTOT + (size_t)stream * N1) * Dm;
  const float* Wp = stream ? w_s1 : w_s0;
  const float* bp = stream ? b_s1 : b_s0;
  const float* ct = stream ? cosT2 : cosT1;
  const float* st = stream ? sinT2 : sinT1;

  v8f acc[2][4] = {};

  const int ar = t >> 1, ak = (t & 1) * 16;   // A stage: 128 rows x 32 k
  const int bk = t >> 3, bc = (t & 7) * 16;   // B stage: 32 k x 128 cols

  for (int kt = 0; kt < 64; ++kt) {
    __syncthreads();
    // ---- stage A tile ----
    if (EPI < 3) {
      const float* src = Ap + (size_t)(rowBase + ar) * Dm + kt * 32 + ak;
      #pragma unroll
      for (int j = 0; j < 4; ++j) {
        float4 f = *(const float4*)(src + 4 * j);
        As[ar][ak + 4*j + 0] = f2bf(f.x);
        As[ar][ak + 4*j + 1] = f2bf(f.y);
        As[ar][ak + 4*j + 2] = f2bf(f.z);
        As[ar][ak + 4*j + 3] = f2bf(f.w);
      }
    } else {
      const uint16_t* src = Ab + (size_t)(rowBase + ar) * Dm + kt * 32 + ak;
      copy16_g2l(src,     &As[ar][ak]);       // async global->LDS (ASYNCcnt)
      copy16_g2l(src + 8, &As[ar][ak + 8]);
    }
    // ---- stage B tile transposed: Bs[col][k] ----
    {
      const float* src = Wp + (size_t)(kt * 32 + bk) * Dm + colBase + bc;
      #pragma unroll
      for (int j = 0; j < 4; ++j) {
        float4 f = *(const float4*)(src + 4 * j);
        Bs[bc + 4*j + 0][bk] = f2bf(f.x);
        Bs[bc + 4*j + 1][bk] = f2bf(f.y);
        Bs[bc + 4*j + 2][bk] = f2bf(f.z);
        Bs[bc + 4*j + 3][bk] = f2bf(f.w);
      }
      if (kt + 1 < 64)
        __builtin_prefetch(Wp + (size_t)((kt + 1) * 32 + bk) * Dm + colBase + bc, 0, 1);
    }
    if (EPI == 3) wait_async_done();
    __syncthreads();

    // ---- fragments (documented 16-bit A/B lane swizzles) ----
    Frag16 fa[2], fb[4];
    #pragma unroll
    for (int mi = 0; mi < 2; ++mi) {
      const uint16_t* p = &As[wr + mi * 16 + l15][half * 8];
      #pragma unroll
      for (int j = 0; j < 4; ++j) {
        fa[mi].u[j]     = *(const uint32_t*)(p + 2 * j);        // K pairs 0,2,4,6 (+half*8)
        fa[mi].u[4 + j] = *(const uint32_t*)(p + 16 + 2 * j);   // K pairs 16..22 (+half*8)
      }
    }
    #pragma unroll
    for (int ni = 0; ni < 4; ++ni) {
      const uint16_t* p = &Bs[wc + ni * 16 + l15][half * 8];    // lanes 0-15: K0-7,16-23; 16-31: K8-15,24-31
      fb[ni].q[0] = *(const uint4*)(p);
      fb[ni].q[1] = *(const uint4*)(p + 16);
    }
    #pragma unroll
    for (int mi = 0; mi < 2; ++mi)
      #pragma unroll
      for (int ni = 0; ni < 4; ++ni)
        acc[mi][ni] = WMMA_BF16(fa[mi].v, fb[ni].v, acc[mi][ni]);
  }

  // ---- epilogue: bias (+scale, +RoPE) + store ----
  #pragma unroll
  for (int mi = 0; mi < 2; ++mi) {
    #pragma unroll
    for (int ni = 0; ni < 4; ++ni) {
      const int col = colBase + wc + ni * 16 + l15;
      const float bv = bp[col];
      #pragma unroll
      for (int v = 0; v < 8; ++v) {
        const int r = rowBase + wr + mi * 16 + v + half * 8;    // row within this 1024-row GEMM
        float val = acc[mi][ni][v] + bv;
        if (EPI == 0) val *= 0.08838834764831843f;              // DH^-0.5
        if (EPI == 0 || EPI == 1) {
          const int dh = col & (DH - 1);
          const float cc = ct[(size_t)r * DH + dh];
          const float ss = st[(size_t)r * DH + dh];
          const float partner = __shfl_xor(val, 1, 32);         // interleaved pair lives in lane^1
          const float rot = (lane & 1) ? partner : -partner;
          val = val * cc + rot * ss;
          const int h = col >> 7;
          const int pos = stream * N1 + r;
          uint16_t* dst = (EPI == 0) ? out_q : out_k;
          dst[(((size_t)(b * Hh + h)) * NTOT + pos) * DH + dh] = f2bf(val);
        } else if (EPI == 2) {
          const int h = col >> 7, dh = col & (DH - 1);
          const int pos = stream * N1 + r;
          out_vt[(((size_t)(b * Hh + h)) * DH + dh) * NTOT + pos] = f2bf(val);
        } else {
          out_f[(size_t)stream * Bv * N1 * Dm + ((size_t)b * N1 + r) * Dm + col] = val;
        }
      }
    }
  }
}

// ---------------------------------------------------------------------------
// Flash attention over the joint N=2048 stream.
// Block = 256 thr (8 waves) = one (b,h), 128 query rows; each wave owns a
// 16-row query tile. Per 32-kv chunk the block cooperatively async-stages the
// 8KB K tile and 8KB V tile into LDS (once, instead of 8x per-wave global
// loads), then every wave runs scores/softmax/PV out of LDS.
// ---------------------------------------------------------------------------
__global__ __launch_bounds__(256)
void attn_kernel(const uint16_t* __restrict__ Qbf, const uint16_t* __restrict__ Kbf,
                 const uint16_t* __restrict__ Vtbf, uint16_t* __restrict__ Obf)
{
  __shared__ uint16_t Ks[32][136];     // kv-chunk K tile [kv][dh]   (stride 68 dwords)
  __shared__ uint16_t Vs[128][40];     // kv-chunk V tile [dh][kv]   (stride 20 dwords)
  __shared__ uint16_t Pl[8][16][34];   // per-wave P-tile re-swizzle scratch

  const int t = threadIdx.x, wid = t >> 5, lane = t & 31, half = lane >> 4, l15 = lane & 15;
  const int bh = blockIdx.y;
  const int b = bh >> 4, h = bh & 15;
  const int qbase = blockIdx.x * 128 + wid * 16;

  const uint16_t* Qh = Qbf  + (size_t)bh * NTOT * DH;
  const uint16_t* Kh = Kbf  + (size_t)bh * NTOT * DH;
  const uint16_t* Vh = Vtbf + (size_t)bh * DH * NTOT;

  // Q A-fragments resident for all kv chunks (4 K-steps of 32 over DH=128)
  Frag16 aq[4];
  {
    const uint16_t* p = Qh + (size_t)(qbase + l15) * DH + half * 8;
    #pragma unroll
    for (int s = 0; s < 4; ++s) {
      #pragma unroll
      for (int j = 0; j < 4; ++j) {
        aq[s].u[j]     = *(const uint32_t*)(p + s * 32 + 2 * j);
        aq[s].u[4 + j] = *(const uint32_t*)(p + s * 32 + 16 + 2 * j);
      }
    }
  }

  float m[8], l[8];
  v8f acc[8] = {};
  #pragma unroll
  for (int v = 0; v < 8; ++v) { m[v] = -3.0e38f; l[v] = 0.f; }

  uint16_t* pl = &Pl[wid][0][0];

  for (int c = 0; c < 64; ++c) {
    const int kvb = c * 32;

    // ---- cooperative async staging of K (32x128) and V (128x32) tiles ----
    __syncthreads();                       // previous chunk's LDS reads done
    #pragma unroll
    for (int p = 0; p < 2; ++p) {
      const int ck = p * 256 + t;          // 512 x 16B chunks for K
      const int krow = ck >> 4, kcol = (ck & 15) * 8;
      copy16_g2l(Kh + (size_t)(kvb + krow) * DH + kcol, &Ks[krow][kcol]);
      const int vrow = ck >> 2, vcol = (ck & 3) * 8;   // 512 x 16B chunks for V
      copy16_g2l(Vh + (size_t)vrow * NTOT + kvb + vcol, &Vs[vrow][vcol]);
    }
    wait_async_done();
    __syncthreads();

    // ---- scores: two 16x16 tiles, K = DH = 128 (4 WMMAs each) ----
    v8f s0 = {}, s1 = {};
    #pragma unroll
    for (int s = 0; s < 4; ++s) {
      Frag16 kb0, kb1;
      const uint16_t* p0 = &Ks[l15][s * 32 + half * 8];
      const uint16_t* p1 = &Ks[16 + l15][s * 32 + half * 8];
      kb0.q[0] = *(const uint4*)(p0);  kb0.q[1] = *(const uint4*)(p0 + 16);
      kb1.q[0] = *(const uint4*)(p1);  kb1.q[1] = *(const uint4*)(p1 + 16);
      s0 = WMMA_BF16(aq[s].v, kb0.v, s0);
      s1 = WMMA_BF16(aq[s].v, kb1.v, s1);
    }

    // ---- online softmax (per row = per VGPR slot, reduce inside 16-lane half) ----
    float alpha[8];
    #pragma unroll
    for (int v = 0; v < 8; ++v) {
      const float a0 = s0[v], a1 = s1[v];
      float mx = fmaxf(a0, a1);
      #pragma unroll
      for (int d = 1; d < 16; d <<= 1) mx = fmaxf(mx, __shfl_xor(mx, d, 32));
      const float mnew = fmaxf(m[v], mx);
      const float al = __expf(m[v] - mnew);
      m[v] = mnew; alpha[v] = al;
      const float p0 = __expf(a0 - mnew);
      const float p1 = __expf(a1 - mnew);
      float ps = p0 + p1;
      #pragma unroll
      for (int d = 1; d < 16; d <<= 1) ps += __shfl_xor(ps, d, 32);
      l[v] = l[v] * al + ps;
      s0[v] = p0; s1[v] = p1;
    }
    #pragma unroll
    for (int dt = 0; dt < 8; ++dt)
      #pragma unroll
      for (int v = 0; v < 8; ++v) acc[dt][v] *= alpha[v];

    // ---- P: C-layout -> LDS row-major -> A-layout fragment (wave-local) ----
    #pragma unroll
    for (int v = 0; v < 8; ++v) {
      const int r = v + half * 8;
      pl[r * 34 + l15]      = f2bf(s0[v]);
      pl[r * 34 + 16 + l15] = f2bf(s1[v]);
    }
    Frag16 fa;
    {
      const uint16_t* p = pl + l15 * 34 + half * 8;
      #pragma unroll
      for (int j = 0; j < 4; ++j) {
        fa.u[j]     = *(const uint32_t*)(p + 2 * j);
        fa.u[4 + j] = *(const uint32_t*)(p + 16 + 2 * j);
      }
    }

    // ---- O += P x V  (V tile [dh][kv] in LDS -> contiguous B-fragments) ----
    #pragma unroll
    for (int dt = 0; dt < 8; ++dt) {
      Frag16 vb;
      const uint16_t* p = &Vs[dt * 16 + l15][half * 8];
      vb.q[0] = *(const uint4*)(p);
      vb.q[1] = *(const uint4*)(p + 16);
      acc[dt] = WMMA_BF16(fa.v, vb.v, acc[dt]);
    }
  }

  // ---- finalize: divide by row sum, store bf16 O [B, N, D] ----
  #pragma unroll
  for (int dt = 0; dt < 8; ++dt) {
    #pragma unroll
    for (int v = 0; v < 8; ++v) {
      const float o = acc[dt][v] / l[v];
      const int n   = qbase + v + half * 8;
      const int col = h * DH + dt * 16 + l15;
      Obf[((size_t)b * NTOT + n) * Dm + col] = f2bf(o);
    }
  }
}

// ---------------------------------------------------------------------------
extern "C" void kernel_launch(void* const* d_in, const int* in_sizes, int n_in,
                              void* d_out, int out_size, void* d_ws, size_t ws_size,
                              hipStream_t stream)
{
  const float* x1   = (const float*)d_in[0];
  const float* x2   = (const float*)d_in[1];
  const float* cos1 = (const float*)d_in[2];
  const float* sin1 = (const float*)d_in[3];
  const float* cos2 = (const float*)d_in[4];
  const float* sin2 = (const float*)d_in[5];
  const float* wq1 = (const float*)d_in[6];  const float* bq1 = (const float*)d_in[7];
  const float* wk1 = (const float*)d_in[8];  const float* bk1 = (const float*)d_in[9];
  const float* wv1 = (const float*)d_in[10]; const float* bv1 = (const float*)d_in[11];
  const float* wo1 = (const float*)d_in[12]; const float* bo1 = (const float*)d_in[13];
  const float* wq2 = (const float*)d_in[14]; const float* bq2 = (const float*)d_in[15];
  const float* wk2 = (const float*)d_in[16]; const float* bk2 = (const float*)d_in[17];
  const float* wv2 = (const float*)d_in[18]; const float* bv2 = (const float*)d_in[19];
  const float* wo2 = (const float*)d_in[20]; const float* bo2 = (const float*)d_in[21];
  float* out = (float*)d_out;

  const size_t per = (size_t)Bv * Hh * NTOT * DH;     // 8,388,608 bf16 elements (16 MB)
  uint16_t* Qbf = (uint16_t*)d_ws;
  uint16_t* Kbf = Qbf + per;
  uint16_t* Vt  = Kbf + per;
  uint16_t* Obf = Vt  + per;                          // total 64 MB workspace

  dim3 blk(256), gGemm(128, 4), gAttn(16, 32);

  gemm_epi_kernel<0><<<gGemm, blk, 0, stream>>>(x1, x2, nullptr, wq1, wq2, bq1, bq2,
                                                cos1, sin1, cos2, sin2,
                                                Qbf, nullptr, nullptr, nullptr);
  gemm_epi_kernel<1><<<gGemm, blk, 0, stream>>>(x1, x2, nullptr, wk1, wk2, bk1, bk2,
                                                cos1, sin1, cos2, sin2,
                                                nullptr, Kbf, nullptr, nullptr);
  gemm_epi_kernel<2><<<gGemm, blk, 0, stream>>>(x1, x2, nullptr, wv1, wv2, bv1, bv2,
                                                cos1, sin1, cos2, sin2,
                                                nullptr, nullptr, Vt, nullptr);
  attn_kernel<<<gAttn, blk, 0, stream>>>(Qbf, Kbf, Vt, Obf);
  gemm_epi_kernel<3><<<gGemm, blk, 0, stream>>>(nullptr, nullptr, Obf, wo1, wo2, bo1, bo2,
                                                cos1, sin1, cos2, sin2,
                                                nullptr, nullptr, nullptr, out);

  (void)in_sizes; (void)n_in; (void)out_size; (void)ws_size;
}
